// MoeBase_40295383171123
// MI455X (gfx1250) — matrix operand order repaired and checked by reading, not compiled, not run on hardware
//
#include <hip/hip_runtime.h>
#include <hip/hip_bf16.h>
#include <math.h>

// Problem sizes (fixed by the reference)
#define T_TOK 8192   // B*S tokens
#define DIM   1024   // D
#define NEXP  8      // E experts
#define HID   4096   // H hidden

typedef __attribute__((ext_vector_type(16))) __bf16 v16bf;
typedef __attribute__((ext_vector_type(8)))  __bf16 v8bf;
typedef __attribute__((ext_vector_type(8)))  float  v8f;

__device__ __forceinline__ float softplus_f(float v) {
    return v > 20.f ? v : log1pf(__expf(v));
}

// ---------------------------------------------------------------------------
// Kernel 1: NoisyTopK router. One wave (32 lanes) per token.
// ---------------------------------------------------------------------------
__global__ __launch_bounds__(256)
void moe_router(const float* __restrict__ x, const float* __restrict__ noise,
                const float* __restrict__ Wg, const float* __restrict__ bg,
                const float* __restrict__ Wn, const float* __restrict__ bn,
                int* __restrict__ topkIdx, float* __restrict__ topkGate)
{
    const int t    = blockIdx.x * 8 + (threadIdx.x >> 5);
    const int lane = threadIdx.x & 31;
    const float* xr = x + (size_t)t * DIM;

    float aG[NEXP], aN[NEXP];
    #pragma unroll
    for (int e = 0; e < NEXP; ++e) { aG[e] = 0.f; aN[e] = 0.f; }

    for (int d0 = lane * 4; d0 < DIM; d0 += 128) {
        const float4 xv = *(const float4*)(xr + d0);
        const float xs[4] = {xv.x, xv.y, xv.z, xv.w};
        #pragma unroll
        for (int j = 0; j < 4; ++j) {
            const float* wg = Wg + (size_t)(d0 + j) * NEXP;
            const float* wn = Wn + (size_t)(d0 + j) * NEXP;
            #pragma unroll
            for (int e = 0; e < NEXP; ++e) {
                aG[e] = fmaf(xs[j], wg[e], aG[e]);
                aN[e] = fmaf(xs[j], wn[e], aN[e]);
            }
        }
    }
    #pragma unroll
    for (int off = 16; off > 0; off >>= 1) {
        #pragma unroll
        for (int e = 0; e < NEXP; ++e) {
            aG[e] += __shfl_xor(aG[e], off, 32);
            aN[e] += __shfl_xor(aN[e], off, 32);
        }
    }
    if (lane == 0) {
        float ny[NEXP];
        #pragma unroll
        for (int e = 0; e < NEXP; ++e) {
            const float sp = softplus_f(aN[e] + bn[e]);
            ny[e] = aG[e] + bg[e] + noise[(size_t)t * NEXP + e] * sp;
        }
        int i0 = 0;
        #pragma unroll
        for (int e = 1; e < NEXP; ++e) if (ny[e] > ny[i0]) i0 = e;
        int i1 = (i0 == 0) ? 1 : 0;
        #pragma unroll
        for (int e = 0; e < NEXP; ++e) if (e != i0 && ny[e] > ny[i1]) i1 = e;
        const float ex = __expf(ny[i1] - ny[i0]);
        const float g0 = 1.f / (1.f + ex);
        topkIdx [t * 2 + 0] = i0;
        topkIdx [t * 2 + 1] = i1;
        topkGate[t * 2 + 0] = g0;
        topkGate[t * 2 + 1] = 1.f - g0;
    }
}

// ---------------------------------------------------------------------------
// Kernel 2: deterministic compaction (ballot/popcount ordered scan).
// ---------------------------------------------------------------------------
__global__ __launch_bounds__(256)
void moe_compact(const int* __restrict__ topkIdx, const float* __restrict__ topkGate,
                 int* __restrict__ counts, int* __restrict__ hbase,
                 int* __restrict__ lists, float* __restrict__ gates)
{
    const int e    = threadIdx.x >> 5;   // wave index == expert id (8 waves)
    const int lane = threadIdx.x & 31;
    int cnt = 0;
    for (int t0 = 0; t0 < T_TOK; t0 += 32) {
        const int t  = t0 + lane;
        const int i0 = topkIdx[t * 2 + 0];
        const int i1 = topkIdx[t * 2 + 1];
        const bool s0 = (i0 == e), s1 = (i1 == e);
        const bool sel = s0 || s1;
        const float g = s0 ? topkGate[t * 2 + 0] : topkGate[t * 2 + 1];
        const unsigned m32 = (unsigned)__ballot(sel);
        if (sel) {
            const int rank = cnt + __popc(m32 & ((1u << lane) - 1u));
            lists[e * T_TOK + rank] = t;
            gates[e * T_TOK + rank] = g;
        }
        cnt += __popc(m32);
    }
    if (lane == 0) counts[e] = cnt;
    __syncthreads();
    if (threadIdx.x == 0) {
        int base = 0;
        for (int i = 0; i < NEXP; ++i) {
            hbase[i] = base;
            base += (counts[i] + 127) & ~127;   // pad to 128-row tile granularity
        }
    }
}

// ---------------------------------------------------------------------------
// WMMA fragment loaders (wave32, 16x16x32 bf16; layouts per CDNA5 ISA 7.12.2)
// ---------------------------------------------------------------------------
__device__ __forceinline__ v16bf load_afrag(const __bf16* lA, int mt, int lane) {
    const int r = lane & 15, half = lane >> 4;
    const v8bf lo = *(const v8bf*)&lA[(mt * 16 + r) * 32 + half * 8];
    const v8bf hi = *(const v8bf*)&lA[(mt * 16 + r) * 32 + half * 8 + 16];
    return __builtin_shufflevector(lo, hi, 0,1,2,3,4,5,6,7,8,9,10,11,12,13,14,15);
}
__device__ __forceinline__ v16bf load_bfrag(const __bf16* lB, int nt, int lane) {
    const int c = lane & 15, kh = lane >> 4;
    const v8bf lo = *(const v8bf*)&lB[(nt * 16 + c) * 32 + kh * 8];
    const v8bf hi = *(const v8bf*)&lB[(nt * 16 + c) * 32 + kh * 8 + 16];
    return __builtin_shufflevector(lo, hi, 0,1,2,3,4,5,6,7,8,9,10,11,12,13,14,15);
}

// Convert + commit one 8-float A chunk into [row][k] LDS tile.
__device__ __forceinline__ void commitA(__bf16* lA, float4 a0, float4 a1,
                                        int arow, int akg) {
    __bf16* p = &lA[arow * 32 + akg];
    p[0] = (__bf16)a0.x; p[1] = (__bf16)a0.y; p[2] = (__bf16)a0.z; p[3] = (__bf16)a0.w;
    p[4] = (__bf16)a1.x; p[5] = (__bf16)a1.y; p[6] = (__bf16)a1.z; p[7] = (__bf16)a1.w;
}
// Convert + commit one 8-float B chunk TRANSPOSED into [n][k] LDS tile.
__device__ __forceinline__ void commitB(__bf16* lB, float4 b0, float4 b1,
                                        int bk, int bng) {
    lB[(bng + 0) * 32 + bk] = (__bf16)b0.x;
    lB[(bng + 1) * 32 + bk] = (__bf16)b0.y;
    lB[(bng + 2) * 32 + bk] = (__bf16)b0.z;
    lB[(bng + 3) * 32 + bk] = (__bf16)b0.w;
    lB[(bng + 4) * 32 + bk] = (__bf16)b1.x;
    lB[(bng + 5) * 32 + bk] = (__bf16)b1.y;
    lB[(bng + 6) * 32 + bk] = (__bf16)b1.z;
    lB[(bng + 7) * 32 + bk] = (__bf16)b1.w;
}

// CDNA5 async global->LDS copy (16B), tracked by ASYNCcnt.
__device__ __forceinline__ void async_copy_b128(__bf16* lds_dst, const __bf16* gsrc) {
    const unsigned lds_off = (unsigned)(size_t)lds_dst;
    const unsigned long long ga = (unsigned long long)(size_t)gsrc;
    asm volatile("global_load_async_to_lds_b128 %0, %1, off"
                 :: "v"(lds_off), "v"(ga) : "memory");
}
__device__ __forceinline__ void wait_async0() {
    asm volatile("s_wait_asynccnt 0" ::: "memory");
}

// ---------------------------------------------------------------------------
// Kernel 3: h = relu(x_gathered @ W1[e] + b1[e]) -> bf16 hbuf
// 512 thr = 16 waves in 4x4 grid; block tile 128(M)x128(N); wave tile 32x32
// (2x2 WMMA register blocking); LDS double-buffered, one barrier per K-step.
// Global loads for step k+1 are issued BEFORE the WMMA block so their latency
// hides under the matrix ops; convert+LDS-commit happens after.
// ---------------------------------------------------------------------------
__global__ __launch_bounds__(512)
void moe_ffn1(const float* __restrict__ x, const float* __restrict__ W1,
              const float* __restrict__ b1,
              const int* __restrict__ counts, const int* __restrict__ hbase,
              const int* __restrict__ lists, __bf16* __restrict__ hbuf)
{
    const int e   = blockIdx.z;
    const int cnt = counts[e];
    const int m0  = blockIdx.y * 128;
    if (m0 >= cnt) return;
    const int n0  = blockIdx.x * 128;      // H-column base

    __shared__ __bf16 lA[2][128 * 32];     // [m][k]
    __shared__ __bf16 lB[2][128 * 32];     // [n][k] (transposed)
    __shared__ int    sIdx[128];

    const int tid = threadIdx.x;
    if (tid < 128) {
        int gr = m0 + tid;
        if (gr >= cnt) gr = cnt - 1;
        sIdx[tid] = lists[e * T_TOK + gr];
    }
    __syncthreads();

    const float* W1e = W1 + (size_t)e * DIM * HID;
    const int lane = tid & 31, w = tid >> 5;
    const int wm = w & 3, wn = w >> 2;     // 4x4 wave grid
    v8f acc00 = {}, acc01 = {}, acc10 = {}, acc11 = {};

    const int arow = tid >> 2, akg = (tid & 3) * 8;   // A staging: 8 floats/thread
    const int bk   = tid >> 4, bng = (tid & 15) * 8;  // B staging: 8 floats/thread
    const float* aptr = x + (size_t)sIdx[arow] * DIM + akg;
    const float* bptr = W1e + (size_t)bk * HID + n0 + bng;

    // prologue: stage k0 = 0 into buffer 0
    {
        const float4 a0 = *(const float4*)(aptr);
        const float4 a1 = *(const float4*)(aptr + 4);
        const float4 b0 = *(const float4*)(bptr);
        const float4 b1 = *(const float4*)(bptr + 4);
        commitA(lA[0], a0, a1, arow, akg);
        commitB(lB[0], b0, b1, bk, bng);
    }
    __syncthreads();

    int cur = 0;
    for (int k0 = 0; k0 < DIM; k0 += 32) {
        const bool more = (k0 + 32) < DIM;
        float4 a0, a1, bq0, bq1;
        if (more) {   // issue next-tile global loads early
            a0  = *(const float4*)(aptr + (k0 + 32));
            a1  = *(const float4*)(aptr + (k0 + 32) + 4);
            bq0 = *(const float4*)(bptr + (size_t)(k0 + 32) * HID);
            bq1 = *(const float4*)(bptr + (size_t)(k0 + 32) * HID + 4);
            __builtin_prefetch(bptr + (size_t)(k0 + 64) * HID, 0, 3);
        }
        const v16bf fa0 = load_afrag(lA[cur], wm * 2 + 0, lane);
        const v16bf fa1 = load_afrag(lA[cur], wm * 2 + 1, lane);
        const v16bf fb0 = load_bfrag(lB[cur], wn * 2 + 0, lane);
        const v16bf fb1 = load_bfrag(lB[cur], wn * 2 + 1, lane);
        acc00 = __builtin_amdgcn_wmma_f32_16x16x32_bf16(false, fa0, false, fb0, (short)0, acc00, false, false);
        acc01 = __builtin_amdgcn_wmma_f32_16x16x32_bf16(false, fa0, false, fb1, (short)0, acc01, false, false);
        acc10 = __builtin_amdgcn_wmma_f32_16x16x32_bf16(false, fa1, false, fb0, (short)0, acc10, false, false);
        acc11 = __builtin_amdgcn_wmma_f32_16x16x32_bf16(false, fa1, false, fb1, (short)0, acc11, false, false);
        if (more) {
            commitA(lA[cur ^ 1], a0, a1, arow, akg);
            commitB(lB[cur ^ 1], bq0, bq1, bk, bng);
        }
        __syncthreads();
        cur ^= 1;
    }

    const int half = lane >> 4;
    const int hb   = hbase[e];
    const v8f* accs[4] = {&acc00, &acc01, &acc10, &acc11};
    #pragma unroll
    for (int mi = 0; mi < 2; ++mi) {
        #pragma unroll
        for (int ni = 0; ni < 2; ++ni) {
            const v8f a = *accs[mi * 2 + ni];
            const int cg = n0 + (wn * 2 + ni) * 16 + (lane & 15);
            const float bias = b1[(size_t)e * HID + cg];
            #pragma unroll
            for (int v = 0; v < 8; ++v) {
                const int gr = m0 + (wm * 2 + mi) * 16 + v + half * 8;
                if (gr < cnt) {
                    float hv = a[v] + bias;
                    hv = hv > 0.f ? hv : 0.f;
                    hbuf[(size_t)(hb + gr) * HID + cg] = (__bf16)hv;
                }
            }
        }
    }
}

// ---------------------------------------------------------------------------
// Kernel 4: out += gate * (h @ W2[e] + b2[e]).  A operand is already bf16 in
// memory -> moved with CDNA5 async global->LDS copies (ASYNCcnt).  Async copy
// for step k+1 is ISSUED before the WMMA block; the s_wait_asynccnt sits just
// before the barrier so copy latency overlaps the matrix ops.
// ---------------------------------------------------------------------------
__global__ __launch_bounds__(512)
void moe_ffn2(const __bf16* __restrict__ hbuf, const float* __restrict__ W2,
              const float* __restrict__ b2,
              const int* __restrict__ counts, const int* __restrict__ hbase,
              const int* __restrict__ lists, const float* __restrict__ gates,
              float* __restrict__ out)
{
    const int e   = blockIdx.z;
    const int cnt = counts[e];
    const int m0  = blockIdx.y * 128;
    if (m0 >= cnt) return;
    const int n0  = blockIdx.x * 128;      // D-column base

    __shared__ __bf16 lA[2][128 * 32];
    __shared__ __bf16 lB[2][128 * 32];
    __shared__ int    sIdx[128];
    __shared__ float  sGate[128];

    const int tid = threadIdx.x;
    if (tid < 128) {
        int gr = m0 + tid;
        const bool valid = gr < cnt;
        if (!valid) gr = cnt - 1;
        sIdx[tid]  = lists[e * T_TOK + gr];
        sGate[tid] = valid ? gates[e * T_TOK + gr] : 0.f;
    }

    const float* W2e = W2 + (size_t)e * HID * DIM;
    const int lane = tid & 31, w = tid >> 5;
    const int wm = w & 3, wn = w >> 2;
    v8f acc00 = {}, acc01 = {}, acc10 = {}, acc11 = {};

    const int arow = tid >> 2, akg = (tid & 3) * 8;   // 8 bf16 = 16B per thread
    const int bk   = tid >> 4, bng = (tid & 15) * 8;
    const __bf16* aptr = hbuf + (size_t)(hbase[e] + m0 + arow) * HID + akg;
    const float*  bptr = W2e + (size_t)bk * DIM + n0 + bng;

    // prologue: async A tile + manual B tile into buffer 0
    async_copy_b128(&lA[0][arow * 32 + akg], aptr);
    {
        const float4 b0 = *(const float4*)(bptr);
        const float4 b1 = *(const float4*)(bptr + 4);
        commitB(lB[0], b0, b1, bk, bng);
    }
    wait_async0();
    __syncthreads();

    int cur = 0;
    for (int k0 = 0; k0 < HID; k0 += 32) {
        const bool more = (k0 + 32) < HID;
        float4 bq0, bq1;
        if (more) {   // issue next-tile copies early: async A + global B loads
            async_copy_b128(&lA[cur ^ 1][arow * 32 + akg], aptr + (k0 + 32));
            bq0 = *(const float4*)(bptr + (size_t)(k0 + 32) * DIM);
            bq1 = *(const float4*)(bptr + (size_t)(k0 + 32) * DIM + 4);
            __builtin_prefetch(bptr + (size_t)(k0 + 64) * DIM, 0, 3);
        }
        const v16bf fa0 = load_afrag(lA[cur], wm * 2 + 0, lane);
        const v16bf fa1 = load_afrag(lA[cur], wm * 2 + 1, lane);
        const v16bf fb0 = load_bfrag(lB[cur], wn * 2 + 0, lane);
        const v16bf fb1 = load_bfrag(lB[cur], wn * 2 + 1, lane);
        acc00 = __builtin_amdgcn_wmma_f32_16x16x32_bf16(false, fa0, false, fb0, (short)0, acc00, false, false);
        acc01 = __builtin_amdgcn_wmma_f32_16x16x32_bf16(false, fa0, false, fb1, (short)0, acc01, false, false);
        acc10 = __builtin_amdgcn_wmma_f32_16x16x32_bf16(false, fa1, false, fb0, (short)0, acc10, false, false);
        acc11 = __builtin_amdgcn_wmma_f32_16x16x32_bf16(false, fa1, false, fb1, (short)0, acc11, false, false);
        if (more) {
            commitB(lB[cur ^ 1], bq0, bq1, bk, bng);
        }
        wait_async0();   // drain async A copy before publishing the buffer
        __syncthreads();
        cur ^= 1;
    }

    const int half = lane >> 4;
    const v8f* accs[4] = {&acc00, &acc01, &acc10, &acc11};
    #pragma unroll
    for (int mi = 0; mi < 2; ++mi) {
        #pragma unroll
        for (int ni = 0; ni < 2; ++ni) {
            const v8f a = *accs[mi * 2 + ni];
            const int cg = n0 + (wn * 2 + ni) * 16 + (lane & 15);
            const float bias = b2[(size_t)e * DIM + cg];
            #pragma unroll
            for (int v = 0; v < 8; ++v) {
                const int rt = (wm * 2 + mi) * 16 + v + half * 8;
                const int gr = m0 + rt;
                if (gr < cnt) {
                    const float y = sGate[rt] * (a[v] + bias);
                    atomicAdd(&out[(size_t)sIdx[rt] * DIM + cg], y);
                }
            }
        }
    }
}

// ---------------------------------------------------------------------------
// Host-side launcher. Workspace (~143 MB):
//   counts[16] | hbase[16] | topkIdx[2T] | topkGate[2T] |
//   lists[E*T] | gates[E*T] | hbuf[(2T + 128E) * H] bf16
// ---------------------------------------------------------------------------
extern "C" void kernel_launch(void* const* d_in, const int* in_sizes, int n_in,
                              void* d_out, int out_size, void* d_ws, size_t ws_size,
                              hipStream_t stream) {
    (void)in_sizes; (void)n_in; (void)out_size; (void)ws_size;
    const float* x     = (const float*)d_in[0];
    const float* noise = (const float*)d_in[1];
    const float* Wg    = (const float*)d_in[2];
    const float* bg    = (const float*)d_in[3];
    const float* Wn    = (const float*)d_in[4];
    const float* bn    = (const float*)d_in[5];
    const float* W1    = (const float*)d_in[6];
    const float* b1    = (const float*)d_in[7];
    const float* W2    = (const float*)d_in[8];
    const float* b2    = (const float*)d_in[9];
    float* out = (float*)d_out;

    char* ws = (char*)d_ws;
    int*    counts   = (int*)ws;    ws += 16 * sizeof(int);
    int*    hbase    = (int*)ws;    ws += 16 * sizeof(int);
    int*    topkIdx  = (int*)ws;    ws += (size_t)T_TOK * 2 * sizeof(int);
    float*  topkGate = (float*)ws;  ws += (size_t)T_TOK * 2 * sizeof(float);
    int*    lists    = (int*)ws;    ws += (size_t)NEXP * T_TOK * sizeof(int);
    float*  gates    = (float*)ws;  ws += (size_t)NEXP * T_TOK * sizeof(float);
    __bf16* hbuf     = (__bf16*)ws; // (2*T_TOK + 128*NEXP) rows x HID bf16

    hipMemsetAsync(d_out, 0, (size_t)T_TOK * DIM * sizeof(float), stream);

    moe_router <<<T_TOK / 8, 256, 0, stream>>>(x, noise, Wg, bg, Wn, bn,
                                               topkIdx, topkGate);
    moe_compact<<<1, 256, 0, stream>>>(topkIdx, topkGate, counts, hbase,
                                       lists, gates);
    moe_ffn1<<<dim3(HID / 128, T_TOK / 128, NEXP), 512, 0, stream>>>(
        x, W1, b1, counts, hbase, lists, hbuf);
    moe_ffn2<<<dim3(DIM / 128, T_TOK / 128, NEXP), 512, 0, stream>>>(
        hbuf, W2, b2, counts, hbase, lists, gates, out);
}
